// WanTransformerBlock_61942018343434
// MI455X (gfx1250) — compile-verified, matrix operand order checked by reading, not run on hardware
//
#include <hip/hip_runtime.h>
#include <hip/hip_bf16.h>

// ---- problem constants ------------------------------------------------------
#define SEQ   2048
#define DIMV  1536
#define HEADS 12
#define DH    128
#define NE    32
#define FF    2048
#define NTOPK 4
#define TTOK  2048          // B*S tokens
#define EPSV  1e-6f

// ---- WMMA types (CDNA5, wave32) --------------------------------------------
typedef __attribute__((ext_vector_type(16))) __bf16 v16bf;
typedef __attribute__((ext_vector_type(8)))  float  v8f;
typedef __attribute__((ext_vector_type(4)))  int    v4i;

__device__ __forceinline__ unsigned short f2bf(float f) {
    unsigned u = __float_as_uint(f);
    unsigned r = (u + 0x7fffu + ((u >> 16) & 1u)) >> 16;   // round-to-nearest-even
    return (unsigned short)r;
}

// Build a 16x32-bf16 A/B fragment from a K-major row.
// ISA 16-bit layout: lane l -> row (l&15); K chunks at kbase=((l>>4)*8) and kbase+16.
__device__ __forceinline__ v16bf load_frag(const unsigned short* rowBase, int lane) {
    union { v16bf v; uint4 q[2]; } f;
    int kb = (lane >> 4) << 3;
    f.q[0] = *(const uint4*)(rowBase + kb);
    f.q[1] = *(const uint4*)(rowBase + kb + 16);
    return f.v;
}

__device__ __forceinline__ v8f wmma_bf16(v16bf a, v16bf b, v8f c) {
    return __builtin_amdgcn_wmma_f32_16x16x32_bf16(false, a, false, b, (short)0, c, false, false);
}

// ---- async global->LDS copy (CDNA5 ASYNCcnt path), guarded fallback ---------
#if __has_builtin(__builtin_amdgcn_global_load_async_to_lds_b128) && \
    __has_builtin(__builtin_amdgcn_s_wait_asynccnt)
#define HAVE_ASYNC_LDS 1
#else
#define HAVE_ASYNC_LDS 0
#endif

__device__ __forceinline__ void cp16(unsigned short* lds, const unsigned short* g) {
#if HAVE_ASYNC_LDS
    __builtin_amdgcn_global_load_async_to_lds_b128(
        (__attribute__((address_space(1))) v4i*)g,
        (__attribute__((address_space(3))) v4i*)lds, 0, 0);
#else
    *(uint4*)lds = *(const uint4*)g;
#endif
}
template<int N> __device__ __forceinline__ void cp_wait() {
#if HAVE_ASYNC_LDS
    __builtin_amdgcn_s_wait_asynccnt(N);
#endif
}

// ---- weight convert+transpose: W[K][N] fp32 -> Wt[N][K] bf16 ---------------
__global__ __launch_bounds__(256) void k_wtrans(const float* __restrict__ W,
                                               unsigned short* __restrict__ Wt,
                                               int K, int N) {
    const float* Wp = W + (size_t)blockIdx.z * K * N;
    unsigned short* Wtp = Wt + (size_t)blockIdx.z * K * N;
    __shared__ unsigned short t[32][33];
    int n0 = blockIdx.x * 32, k0 = blockIdx.y * 32;
    int tx = threadIdx.x & 31, ty = threadIdx.x >> 5;     // 32 x 8
    #pragma unroll
    for (int r = 0; r < 32; r += 8)
        t[ty + r][tx] = f2bf(Wp[(size_t)(k0 + ty + r) * N + n0 + tx]);
    __syncthreads();
    #pragma unroll
    for (int r = 0; r < 32; r += 8)
        Wtp[(size_t)(n0 + ty + r) * K + k0 + tx] = t[tx][ty + r];
}

// ---- tiny kernels -----------------------------------------------------------
__global__ void k_mods(const float* __restrict__ sst, const float* __restrict__ temb,
                       float* __restrict__ mods) {
    int i = blockIdx.x * 256 + threadIdx.x;
    if (i < 6 * DIMV) mods[i] = sst[i] + temb[i];
}

// LayerNorm (no affine) then *(1+scale)+shift; writes bf16 and optionally fp32
__global__ __launch_bounds__(256) void k_ln_mod(const float* __restrict__ x,
                                                const float* __restrict__ mods,
                                                int shiftIdx, int scaleIdx,
                                                unsigned short* __restrict__ hbf,
                                                float* __restrict__ hf) {
    int s = blockIdx.x, tid = threadIdx.x;
    __shared__ float r1[256], r2[256];
    float sm = 0.f, ss = 0.f;
    for (int d = tid; d < DIMV; d += 256) {
        float a = x[(size_t)s * DIMV + d];
        sm += a; ss += a * a;
    }
    r1[tid] = sm; r2[tid] = ss; __syncthreads();
    for (int o = 128; o; o >>= 1) {
        if (tid < o) { r1[tid] += r1[tid + o]; r2[tid] += r2[tid + o]; }
        __syncthreads();
    }
    float mean = r1[0] * (1.f / DIMV);
    float var  = r2[0] * (1.f / DIMV) - mean * mean;
    float rstd = rsqrtf(var + EPSV);
    const float* shift = mods + (size_t)shiftIdx * DIMV;
    const float* scale = mods + (size_t)scaleIdx * DIMV;
    for (int d = tid; d < DIMV; d += 256) {
        float v = (x[(size_t)s * DIMV + d] - mean) * rstd * (1.f + scale[d]) + shift[d];
        hbf[(size_t)s * DIMV + d] = f2bf(v);
        if (hf) hf[(size_t)s * DIMV + d] = v;
    }
}

// ---- unified WMMA GEMM ------------------------------------------------------
// C[M,N] = A(bf16)[M,K] @ Bt(bf16)[N,K]^T, block tile 128x256, BK=32,
// double-buffered LDS with async global->LDS copies (ASYNCcnt pipeline).
// MODE 0: outF = C + bias                         (dense, fp32 out)
// MODE 1: outF = resid + gatevec * (C + bias)     (dense, gated residual)
// MODE 2: outBF = gelu(C)   A rows gathered by slotTok    (MoE up-proj)
// MODE 3: outF  = slotGate * C   A rows = contiguous slots (MoE down-proj)
template<int N, int K, int MODE>
__global__ __launch_bounds__(256) void k_gemm(
    const unsigned short* __restrict__ A,
    const unsigned short* __restrict__ BtG,
    const float* __restrict__ bias,
    float* __restrict__ outF,
    unsigned short* __restrict__ outBF,
    const float* __restrict__ resid,
    const float* __restrict__ gatevec,
    const int* __restrict__ offsets,
    const int* __restrict__ slotTok,
    const float* __restrict__ slotGate) {
    __shared__ unsigned short As[2][128 * 32];   // 16 KB
    __shared__ unsigned short Bs[2][256 * 32];   // 32 KB
    int tid = threadIdx.x, lane = tid & 31, wid = tid >> 5;
    int wm = wid >> 2, wn = wid & 3;             // 2 x 4 wave grid; wave tile 64x64
    int nn = lane & 15, half8 = (lane >> 4) * 8;
    int n0 = blockIdx.x * 256;
    int m0 = blockIdx.y * 128;

    int base = 0, cnt = 0x7fffffff;
    const unsigned short* Bt = BtG;
    if (MODE >= 2) {
        int e = blockIdx.z;
        base = offsets[e];
        cnt = offsets[e + 1] - base;
        if (m0 >= cnt) return;
        Bt += (size_t)e * N * K;
    }

    // staging: A -> thread covers row (tid>>1), 32B half (tid&1); B -> row tid (64B)
    int sr = tid >> 1, sc = (tid & 1) << 4;
    const unsigned short* aRow;
    if (MODE == 2) {
        int lr = m0 + sr; if (lr >= cnt) lr = cnt - 1;      // clamp (rows never stored)
        aRow = A + (size_t)slotTok[base + lr] * K;
    } else if (MODE == 3) {
        int lr = m0 + sr; if (lr >= cnt) lr = cnt - 1;
        aRow = A + (size_t)(base + lr) * K;
    } else {
        aRow = A + (size_t)(m0 + sr) * K;
    }
    const unsigned short* bRow = Bt + (size_t)(n0 + tid) * K;

    v8f z = {};
    v8f acc[4][4];
    #pragma unroll
    for (int i = 0; i < 4; i++)
        #pragma unroll
        for (int j = 0; j < 4; j++) acc[i][j] = z;

    const int NK = K / 32;
    // prologue: stage 0 in flight (6 async copies per thread)
    {
        cp16(&As[0][sr * 32 + sc],     aRow + sc);
        cp16(&As[0][sr * 32 + sc + 8], aRow + sc + 8);
        #pragma unroll
        for (int i = 0; i < 4; i++) cp16(&Bs[0][tid * 32 + i * 8], bRow + i * 8);
    }
    for (int kt = 0; kt < NK; kt++) {
        int buf = kt & 1;
        if (kt + 1 < NK) {
            int kk = (kt + 1) * 32, nb = buf ^ 1;
            cp16(&As[nb][sr * 32 + sc],     aRow + kk + sc);
            cp16(&As[nb][sr * 32 + sc + 8], aRow + kk + sc + 8);
            #pragma unroll
            for (int i = 0; i < 4; i++) cp16(&Bs[nb][tid * 32 + i * 8], bRow + kk + i * 8);
            cp_wait<6>();      // stage kt landed; stage kt+1 still in flight
        } else {
            cp_wait<0>();
        }
        __syncthreads();
        v16bf af[4], bf[4];
        #pragma unroll
        for (int mt = 0; mt < 4; mt++)
            af[mt] = load_frag(&As[buf][(wm * 64 + mt * 16 + nn) * 32], lane);
        #pragma unroll
        for (int nt = 0; nt < 4; nt++)
            bf[nt] = load_frag(&Bs[buf][(wn * 64 + nt * 16 + nn) * 32], lane);
        #pragma unroll
        for (int mt = 0; mt < 4; mt++)
            #pragma unroll
            for (int nt = 0; nt < 4; nt++)
                acc[mt][nt] = wmma_bf16(af[mt], bf[nt], acc[mt][nt]);
        __syncthreads();
    }

    #pragma unroll
    for (int mt = 0; mt < 4; mt++) {
        #pragma unroll
        for (int r = 0; r < 8; r++) {
            int lm = wm * 64 + mt * 16 + half8 + r;
            if (MODE >= 2 && (m0 + lm) >= cnt) continue;
            float gw = (MODE == 3) ? slotGate[base + m0 + lm] : 0.f;
            #pragma unroll
            for (int nt = 0; nt < 4; nt++) {
                int gn = n0 + wn * 64 + nt * 16 + nn;
                float v = acc[mt][nt][r];
                if (MODE == 0) {
                    outF[(size_t)(m0 + lm) * N + gn] = v + bias[gn];
                } else if (MODE == 1) {
                    size_t oi = (size_t)(m0 + lm) * N + gn;
                    outF[oi] = resid[oi] + gatevec[gn] * (v + bias[gn]);
                } else if (MODE == 2) {
                    float g = 0.5f * v * (1.f + erff(v * 0.70710678118654752f));
                    outBF[(size_t)(base + m0 + lm) * N + gn] = f2bf(g);
                } else {
                    outF[(size_t)(base + m0 + lm) * N + gn] = gw * v;
                }
            }
        }
    }
}

// RMS-norm (full DIM) + rotary for q/k; repack per-head bf16; V transposed
__global__ __launch_bounds__(256) void k_qkv_prep(
    const float* __restrict__ qr, const float* __restrict__ kr, const float* __restrict__ vr,
    const float* __restrict__ nqw, const float* __restrict__ nkw, const float* __restrict__ rot,
    unsigned short* __restrict__ qb, unsigned short* __restrict__ kb,
    unsigned short* __restrict__ vT) {
    int s = blockIdx.x, tid = threadIdx.x;
    __shared__ float red[256];
    float sq = 0.f, sk = 0.f;
    for (int d = tid; d < DIMV; d += 256) {
        float a = qr[(size_t)s * DIMV + d]; sq += a * a;
        float b = kr[(size_t)s * DIMV + d]; sk += b * b;
    }
    red[tid] = sq; __syncthreads();
    for (int o = 128; o; o >>= 1) { if (tid < o) red[tid] += red[tid + o]; __syncthreads(); }
    float rq = rsqrtf(red[0] * (1.f / DIMV) + EPSV); __syncthreads();
    red[tid] = sk; __syncthreads();
    for (int o = 128; o; o >>= 1) { if (tid < o) red[tid] += red[tid + o]; __syncthreads(); }
    float rk = rsqrtf(red[0] * (1.f / DIMV) + EPSV);

    for (int p = tid; p < DIMV / 2; p += 256) {
        int d0 = p * 2, d1 = d0 + 1;
        int h = d0 / DH, dd = d0 % DH, fi = dd >> 1;
        float c  = rot[(size_t)s * (DH / 2) + fi];
        float sn = rot[(size_t)SEQ * (DH / 2) + (size_t)s * (DH / 2) + fi];
        float qe = qr[(size_t)s * DIMV + d0] * rq * nqw[d0];
        float qo = qr[(size_t)s * DIMV + d1] * rq * nqw[d1];
        size_t qi = ((size_t)h * SEQ + s) * DH + dd;
        qb[qi]     = f2bf(qe * c - qo * sn);
        qb[qi + 1] = f2bf(qe * sn + qo * c);
        float ke = kr[(size_t)s * DIMV + d0] * rk * nkw[d0];
        float ko = kr[(size_t)s * DIMV + d1] * rk * nkw[d1];
        kb[qi]     = f2bf(ke * c - ko * sn);
        kb[qi + 1] = f2bf(ke * sn + ko * c);
        vT[((size_t)h * DH + dd)     * SEQ + s] = f2bf(vr[(size_t)s * DIMV + d0]);
        vT[((size_t)h * DH + dd + 1) * SEQ + s] = f2bf(vr[(size_t)s * DIMV + d1]);
    }
}

// Flash attention: one wave owns a 16-row Q tile; stream KV in 32-row steps.
__global__ __launch_bounds__(128) void k_attn(
    const unsigned short* __restrict__ qb, const unsigned short* __restrict__ kb,
    const unsigned short* __restrict__ vT, unsigned short* __restrict__ ob) {
    __shared__ unsigned short Ps[4][16 * 32];
    int lane = threadIdx.x & 31, wid = threadIdx.x >> 5;
    int h = blockIdx.y;
    int q0 = (blockIdx.x * 4 + wid) * 16;
    int nn = lane & 15, half8 = (lane >> 4) * 8;
    const float scl = 0.08838834764831845f;   // 1/sqrt(DH)

    v16bf aq[4];
    #pragma unroll
    for (int kd = 0; kd < 4; kd++)
        aq[kd] = load_frag(qb + ((size_t)h * SEQ + q0 + nn) * DH + kd * 32, lane);

    v8f z = {};
    v8f o[8]; for (int i = 0; i < 8; i++) o[i] = z;
    float mo[8], l[8];
    #pragma unroll
    for (int r = 0; r < 8; r++) { mo[r] = -1e30f; l[r] = 0.f; }

    for (int j = 0; j < SEQ / 32; j++) {
        int kv = j * 32;
        v8f s0 = z, s1 = z;
        #pragma unroll
        for (int kd = 0; kd < 4; kd++) {
            v16bf b0 = load_frag(kb + ((size_t)h * SEQ + kv + nn)      * DH + kd * 32, lane);
            v16bf b1 = load_frag(kb + ((size_t)h * SEQ + kv + 16 + nn) * DH + kd * 32, lane);
            s0 = wmma_bf16(aq[kd], b0, s0);
            s1 = wmma_bf16(aq[kd], b1, s1);
        }
        float mx[8];
        #pragma unroll
        for (int r = 0; r < 8; r++) {
            float v0 = s0[r] * scl, v1 = s1[r] * scl;
            s0[r] = v0; s1[r] = v1;
            mx[r] = fmaxf(v0, v1);
        }
        #pragma unroll
        for (int m = 8; m >= 1; m >>= 1)
            #pragma unroll
            for (int r = 0; r < 8; r++) mx[r] = fmaxf(mx[r], __shfl_xor(mx[r], m, 32));
        float alpha[8], rs[8];
        #pragma unroll
        for (int r = 0; r < 8; r++) {
            float mn = fmaxf(mo[r], mx[r]);
            alpha[r] = __expf(mo[r] - mn);
            mo[r] = mn;
            float p0 = __expf(s0[r] - mn), p1 = __expf(s1[r] - mn);
            s0[r] = p0; s1[r] = p1;
            rs[r] = p0 + p1;
        }
        #pragma unroll
        for (int m = 8; m >= 1; m >>= 1)
            #pragma unroll
            for (int r = 0; r < 8; r++) rs[r] += __shfl_xor(rs[r], m, 32);
        #pragma unroll
        for (int r = 0; r < 8; r++) l[r] = l[r] * alpha[r] + rs[r];
        #pragma unroll
        for (int nt = 0; nt < 8; nt++)
            #pragma unroll
            for (int r = 0; r < 8; r++) o[nt][r] *= alpha[r];
        // D-layout P -> LDS in A-layout (row-major 16x32 bf16); wave-private, LDS in-order
        #pragma unroll
        for (int r = 0; r < 8; r++) {
            Ps[wid][(half8 + r) * 32 + nn]      = f2bf(s0[r]);
            Ps[wid][(half8 + r) * 32 + nn + 16] = f2bf(s1[r]);
        }
        v16bf ap = load_frag(&Ps[wid][nn * 32], lane);
        #pragma unroll
        for (int nt = 0; nt < 8; nt++) {
            v16bf bv = load_frag(vT + ((size_t)h * DH + nt * 16 + nn) * SEQ + kv, lane);
            o[nt] = wmma_bf16(ap, bv, o[nt]);
        }
    }
    float inv[8];
    #pragma unroll
    for (int r = 0; r < 8; r++) inv[r] = 1.f / l[r];
    #pragma unroll
    for (int nt = 0; nt < 8; nt++)
        #pragma unroll
        for (int r = 0; r < 8; r++)
            ob[(size_t)(q0 + half8 + r) * DIMV + h * DH + nt * 16 + nn] = f2bf(o[nt][r] * inv[r]);
}

// Router: one wave per token, lane = expert; in-wave top-4 on (sigmoid + bias)
__global__ __launch_bounds__(128) void k_router(
    const float* __restrict__ hf, const float* __restrict__ rw,
    const float* __restrict__ ebias, int* __restrict__ tokExp,
    float* __restrict__ tokW, int* __restrict__ counts) {
    int lane = threadIdx.x & 31, wid = threadIdx.x >> 5;
    int t = blockIdx.x * 4 + wid;
    float acc = 0.f;
    for (int d = 0; d < DIMV; d++)
        acc += hf[(size_t)t * DIMV + d] * rw[(size_t)d * NE + lane];
    float sv  = 1.f / (1.f + __expf(-acc));
    float key = sv + ebias[lane];
    float g[NTOPK]; int id[NTOPK];
    #pragma unroll
    for (int k = 0; k < NTOPK; k++) {
        float v = key; int ii = lane;
        #pragma unroll
        for (int m = 16; m >= 1; m >>= 1) {
            float ov = __shfl_xor(v, m, 32);
            int   oi = __shfl_xor(ii, m, 32);
            if (ov > v || (ov == v && oi < ii)) { v = ov; ii = oi; }
        }
        id[k] = ii; g[k] = __shfl(sv, ii, 32);
        if (lane == ii) key = -1e30f;
    }
    float gs = g[0] + g[1] + g[2] + g[3] + 1e-20f;
    if (lane == 0) {
        #pragma unroll
        for (int k = 0; k < NTOPK; k++) {
            tokExp[t * NTOPK + k] = id[k];
            tokW[t * NTOPK + k]   = g[k] / gs;
            atomicAdd(&counts[id[k]], 1);
        }
    }
}

// Deterministic bucket build: thread e scans all entries, assigns consecutive slots
__global__ void k_route_build(const int* __restrict__ counts, const int* __restrict__ tokExp,
                              const float* __restrict__ tokW, int* __restrict__ offsets,
                              int* __restrict__ slotTok, float* __restrict__ slotGate,
                              int* __restrict__ tokSlot) {
    __shared__ int offs[NE + 1];
    if (threadIdx.x == 0) {
        int a = 0;
        for (int e = 0; e < NE; e++) { offs[e] = a; a += counts[e]; }
        offs[NE] = a;
        for (int e = 0; e <= NE; e++) offsets[e] = offs[e];
    }
    __syncthreads();
    int e = threadIdx.x;
    int pos = offs[e];
    for (int i = 0; i < TTOK * NTOPK; i++) {
        if (tokExp[i] == e) {
            slotTok[pos]  = i >> 2;
            slotGate[pos] = tokW[i];
            tokSlot[i]    = pos;
            pos++;
        }
    }
}

// out = x2 + c_gate * sum_k ypart[tokSlot[t,k]]
__global__ void k_final(const float* __restrict__ x2, const float* __restrict__ ypart,
                        const int* __restrict__ tokSlot, const float* __restrict__ mods,
                        float* __restrict__ out) {
    int i = blockIdx.x * 256 + threadIdx.x;
    if (i >= TTOK * DIMV) return;
    int t = i / DIMV, d = i - t * DIMV;
    float s4 = 0.f;
    #pragma unroll
    for (int k = 0; k < NTOPK; k++)
        s4 += ypart[(size_t)tokSlot[t * NTOPK + k] * DIMV + d];
    out[i] = x2[i] + mods[5 * DIMV + d] * s4;
}

// ---- host launcher ----------------------------------------------------------
extern "C" void kernel_launch(void* const* d_in, const int* in_sizes, int n_in,
                              void* d_out, int out_size, void* d_ws, size_t ws_size,
                              hipStream_t stream) {
    (void)in_sizes; (void)n_in; (void)out_size; (void)ws_size;
    const float* x     = (const float*)d_in[0];
    const float* temb  = (const float*)d_in[2];
    const float* rot   = (const float*)d_in[3];
    const float* sst   = (const float*)d_in[5];
    const float* Wq    = (const float*)d_in[6];
    const float* bq    = (const float*)d_in[7];
    const float* Wk    = (const float*)d_in[8];
    const float* bk    = (const float*)d_in[9];
    const float* Wv    = (const float*)d_in[10];
    const float* bv    = (const float*)d_in[11];
    const float* Wo    = (const float*)d_in[12];
    const float* bo    = (const float*)d_in[13];
    const float* nqw   = (const float*)d_in[14];
    const float* nkw   = (const float*)d_in[15];
    const float* rw    = (const float*)d_in[16];
    const float* ebias = (const float*)d_in[17];
    const float* W1    = (const float*)d_in[18];
    const float* W2    = (const float*)d_in[19];
    float* out = (float*)d_out;

    char* p = (char*)d_ws;
    auto alloc = [&](size_t bytes) -> void* {
        void* r = (void*)p;
        p += (bytes + 255) & ~(size_t)255;
        return r;
    };
    float*          mods     = (float*)alloc((size_t)6 * DIMV * 4);
    unsigned short* h1       = (unsigned short*)alloc((size_t)TTOK * DIMV * 2);
    float*          qraw     = (float*)alloc((size_t)TTOK * DIMV * 4);
    float*          kraw     = (float*)alloc((size_t)TTOK * DIMV * 4);
    float*          vraw     = (float*)alloc((size_t)TTOK * DIMV * 4);
    unsigned short* qb       = (unsigned short*)alloc((size_t)TTOK * DIMV * 2);
    unsigned short* kbuf     = (unsigned short*)alloc((size_t)TTOK * DIMV * 2);
    unsigned short* vT       = (unsigned short*)alloc((size_t)TTOK * DIMV * 2);
    unsigned short* obf      = (unsigned short*)alloc((size_t)TTOK * DIMV * 2);
    float*          x2       = (float*)alloc((size_t)TTOK * DIMV * 4);
    unsigned short* h2       = (unsigned short*)alloc((size_t)TTOK * DIMV * 2);
    float*          h2f      = (float*)alloc((size_t)TTOK * DIMV * 4);
    int*            tokExp   = (int*)alloc((size_t)TTOK * NTOPK * 4);
    float*          tokW     = (float*)alloc((size_t)TTOK * NTOPK * 4);
    int*            counts   = (int*)alloc((size_t)NE * 4);
    int*            offsets  = (int*)alloc((size_t)(NE + 1) * 4);
    int*            slotTok  = (int*)alloc((size_t)TTOK * NTOPK * 4);
    float*          slotGate = (float*)alloc((size_t)TTOK * NTOPK * 4);
    int*            tokSlot  = (int*)alloc((size_t)TTOK * NTOPK * 4);
    unsigned short* act      = (unsigned short*)alloc((size_t)TTOK * NTOPK * FF * 2);
    float*          ypart    = (float*)alloc((size_t)TTOK * NTOPK * DIMV * 4);
    // bf16 transposed weights [N][K]
    unsigned short* Wqt      = (unsigned short*)alloc((size_t)DIMV * DIMV * 2);
    unsigned short* Wkt      = (unsigned short*)alloc((size_t)DIMV * DIMV * 2);
    unsigned short* Wvt      = (unsigned short*)alloc((size_t)DIMV * DIMV * 2);
    unsigned short* Wot      = (unsigned short*)alloc((size_t)DIMV * DIMV * 2);
    unsigned short* W1t      = (unsigned short*)alloc((size_t)NE * DIMV * FF * 2);
    unsigned short* W2t      = (unsigned short*)alloc((size_t)NE * FF * DIMV * 2);

    // weight conversion (fp32 [K][N] -> bf16 [N][K])
    dim3 gtD(DIMV / 32, DIMV / 32, 1);
    k_wtrans<<<gtD, 256, 0, stream>>>(Wq, Wqt, DIMV, DIMV);
    k_wtrans<<<gtD, 256, 0, stream>>>(Wk, Wkt, DIMV, DIMV);
    k_wtrans<<<gtD, 256, 0, stream>>>(Wv, Wvt, DIMV, DIMV);
    k_wtrans<<<gtD, 256, 0, stream>>>(Wo, Wot, DIMV, DIMV);
    k_wtrans<<<dim3(FF / 32, DIMV / 32, NE), 256, 0, stream>>>(W1, W1t, DIMV, FF);
    k_wtrans<<<dim3(DIMV / 32, FF / 32, NE), 256, 0, stream>>>(W2, W2t, FF, DIMV);

    k_mods<<<dim3((6 * DIMV + 255) / 256), 256, 0, stream>>>(sst, temb, mods);
    k_ln_mod<<<TTOK, 256, 0, stream>>>(x, mods, 0, 1, h1, (float*)nullptr);

    dim3 gd(DIMV / 256, TTOK / 128);
    k_gemm<DIMV, DIMV, 0><<<gd, 256, 0, stream>>>(h1, Wqt, bq, qraw, nullptr, nullptr, nullptr,
                                                  nullptr, nullptr, nullptr);
    k_gemm<DIMV, DIMV, 0><<<gd, 256, 0, stream>>>(h1, Wkt, bk, kraw, nullptr, nullptr, nullptr,
                                                  nullptr, nullptr, nullptr);
    k_gemm<DIMV, DIMV, 0><<<gd, 256, 0, stream>>>(h1, Wvt, bv, vraw, nullptr, nullptr, nullptr,
                                                  nullptr, nullptr, nullptr);
    k_qkv_prep<<<TTOK, 256, 0, stream>>>(qraw, kraw, vraw, nqw, nkw, rot, qb, kbuf, vT);

    k_attn<<<dim3(SEQ / 64, HEADS), 128, 0, stream>>>(qb, kbuf, vT, obf);

    // x2 = x + gate * (o @ Wo + bo)
    k_gemm<DIMV, DIMV, 1><<<gd, 256, 0, stream>>>(obf, Wot, bo, x2, nullptr, x, mods + 2 * DIMV,
                                                  nullptr, nullptr, nullptr);
    k_ln_mod<<<TTOK, 256, 0, stream>>>(x2, mods, 3, 4, h2, h2f);

    (void)hipMemsetAsync(counts, 0, NE * sizeof(int), stream);
    k_router<<<TTOK / 4, 128, 0, stream>>>(h2f, rw, ebias, tokExp, tokW, counts);
    k_route_build<<<1, 32, 0, stream>>>(counts, tokExp, tokW, offsets, slotTok, slotGate, tokSlot);

    int maxRT = (TTOK * NTOPK) / 128;   // 64
    k_gemm<FF, DIMV, 2><<<dim3(FF / 256, maxRT, NE), 256, 0, stream>>>(
        h2, W1t, nullptr, nullptr, act, nullptr, nullptr, offsets, slotTok, nullptr);
    k_gemm<DIMV, FF, 3><<<dim3(DIMV / 256, maxRT, NE), 256, 0, stream>>>(
        act, W2t, nullptr, ypart, nullptr, nullptr, nullptr, offsets, nullptr, slotGate);

    k_final<<<(TTOK * DIMV + 255) / 256, 256, 0, stream>>>(x2, ypart, tokSlot, mods, out);
}
